// Decoder_85779086835739
// MI455X (gfx1250) — matrix operand order repaired and checked by reading, not compiled, not run on hardware
//
#include <hip/hip_runtime.h>
#include <hip/hip_bf16.h>

typedef __attribute__((ext_vector_type(16))) _Float16 v16h;
typedef __attribute__((ext_vector_type(8)))  float    v8f;
typedef __attribute__((ext_vector_type(4)))  float    f32x4;

#define Bsz 32
#define Ssz 2048
#define Usz 512
#define Vsz 32000
#define Gsz 2048   // 4*U
#define LDK 72     // LDS row pitch (halfs) for 64-wide K chunks; 144B = 9*16B
#define KSLICES 10 // 32000 = 10 * 3200, 3200 = 50 * 64

__device__ __forceinline__ float sigf(float x) { return 1.0f / (1.0f + __expf(-x)); }

__device__ __forceinline__ v8f wmma32(v16h a, v16h b, v8f c) {
    return __builtin_amdgcn_wmma_f32_16x16x32_f16(false, a, false, b, (short)0, c, false, false);
}

// A fragment (16x32 f16). LDS tile row-major [M][ldk] halfs, ldk % 8 == 0.
__device__ __forceinline__ v16h ldsFragA(const _Float16* base, int ldk, int row0, int kb) {
    int lane = threadIdx.x & 31;
    const _Float16* p = base + (row0 + (lane & 15)) * ldk + kb + ((lane >> 4) << 3);
    v16h f;
    ((f32x4*)&f)[0] = *(const f32x4*)p;
    ((f32x4*)&f)[1] = *(const f32x4*)(p + 16);
    return f;
}
// B fragment (32x16 f16) from row-major [N][ldk] LDS (B stored N-major, K contiguous).
__device__ __forceinline__ v16h ldsFragB(const _Float16* base, int ldk, int row0, int kb) {
    int lane = threadIdx.x & 31;
    const _Float16* p = base + (row0 + (lane & 15)) * ldk + kb + ((lane >> 4) << 3);
    v16h f;
    ((f32x4*)&f)[0] = *(const f32x4*)p;
    ((f32x4*)&f)[1] = *(const f32x4*)(p + 16);
    return f;
}

// ---- register-staged global->LDS f16 conversion, 256 threads, float4 loads ----
template<int ROWS, int KC>
struct StageRegs { float4 t[ROWS * KC / 1024]; };   // (ROWS*KC/4)/256 quads per thread

template<int ROWS, int KC>
__device__ __forceinline__ void loadQ(StageRegs<ROWS, KC>& r, const float* src,
                                      long stride, int k0) {
    constexpr int QPR = KC / 4;
    constexpr int PT  = ROWS * QPR / 256;
#pragma unroll
    for (int i = 0; i < PT; ++i) {
        int idx = i * 256 + (int)threadIdx.x;
        int row = idx / QPR, q = idx % QPR;
        r.t[i] = *(const float4*)(src + (long)row * stride + k0 + q * 4);
    }
}
template<int ROWS, int KC>
__device__ __forceinline__ void storeQ(const StageRegs<ROWS, KC>& r, _Float16* dst, int ldk) {
    constexpr int QPR = KC / 4;
    constexpr int PT  = ROWS * QPR / 256;
#pragma unroll
    for (int i = 0; i < PT; ++i) {
        int idx = i * 256 + (int)threadIdx.x;
        int row = idx / QPR, q = idx % QPR;
        union { _Float16 h[4]; uint2 u; } cv;
        cv.h[0] = (_Float16)r.t[i].x; cv.h[1] = (_Float16)r.t[i].y;
        cv.h[2] = (_Float16)r.t[i].z; cv.h[3] = (_Float16)r.t[i].w;
        *(uint2*)(dst + row * ldk + q * 4) = cv.u;
    }
}

// Pipelined per-wave accumulate: acc += A(32 x kLen) * B(64 x kLen)^T, K-contiguous sources.
// As: 2*(32*LDK) halfs, Bs: 2*(64*LDK). kLen % 64 == 0. 8 waves: (tr=(w>>2)*16, tc=(w&3)*16).
__device__ __forceinline__ void gemm_seg(v8f& acc, _Float16* As, _Float16* Bs,
        const float* aSrc, long aStride, const float* bSrc, long bStride,
        int kStart, int kLen, int tr, int tc) {
    StageRegs<32, 64> ra;
    StageRegs<64, 64> rb;
    loadQ(ra, aSrc, aStride, kStart);
    loadQ(rb, bSrc, bStride, kStart);
    storeQ(ra, As, LDK);
    storeQ(rb, Bs, LDK);
    const int nC = kLen >> 6;
    int cur = 0;
    for (int ci = 0; ci < nC; ++ci) {
        __syncthreads();
        int nxt = cur ^ 1;
        bool more = (ci + 1) < nC;
        if (more) {  // issue next chunk's global loads before computing (latency overlap)
            loadQ(ra, aSrc, aStride, kStart + (ci + 1) * 64);
            loadQ(rb, bSrc, bStride, kStart + (ci + 1) * 64);
        }
        const _Float16* Ab = As + cur * (32 * LDK);
        const _Float16* Bb = Bs + cur * (64 * LDK);
        v16h a0 = ldsFragA(Ab, LDK, tr, 0);
        v16h b0 = ldsFragB(Bb, LDK, tc, 0);
        acc = wmma32(a0, b0, acc);
        v16h a1 = ldsFragA(Ab, LDK, tr, 32);
        v16h b1 = ldsFragB(Bb, LDK, tc, 32);
        acc = wmma32(a1, b1, acc);
        if (more) {
            storeQ(ra, As + nxt * (32 * LDK), LDK);
            storeQ(rb, Bs + nxt * (64 * LDK), LDK);
        }
        cur = nxt;
    }
    __syncthreads();
}

// ---------------- K1: gates = x@W_ih.T (+ h@W_hh.T on slice 0), K-split x10 ----------------
__global__ __launch_bounds__(256) void k_gates(const float* __restrict__ x,
        const float* __restrict__ W_ih, const float* __restrict__ h0,
        const float* __restrict__ W_hh, float* __restrict__ partial) {
    __shared__ _Float16 As[2 * 32 * LDK];
    __shared__ _Float16 Bs[2 * 64 * LDK];
    int tid = threadIdx.x, lane = tid & 31, w = tid >> 5;
    int nt = blockIdx.x;           // n tile (64 wide) over 2048
    int ks = blockIdx.y;           // K slice: [ks*3200, +3200)
    int tr = (w >> 2) * 16, tc = (w & 3) * 16;
    v8f acc = {};
    gemm_seg(acc, As, Bs, x, (long)Vsz, W_ih + (long)nt * 64 * Vsz, (long)Vsz,
             ks * 3200, 3200, tr, tc);
    if (ks == 0) {
        gemm_seg(acc, As, Bs, h0, (long)Usz, W_hh + (long)nt * 64 * Usz, (long)Usz,
                 0, Usz, tr, tc);
    }
    float* outp = partial + (long)ks * (Bsz * Gsz);
    int n  = nt * 64 + tc + (lane & 15);
    int m0 = tr + ((lane >> 4) << 3);
#pragma unroll
    for (int r = 0; r < 8; ++r) outp[(m0 + r) * Gsz + n] = acc[r];
}

// ---------------- K2: LSTM elementwise (sums the K-slice partials) ----------------
__global__ __launch_bounds__(256) void k_lstm(const float* __restrict__ partial,
        const float* __restrict__ c0, const float* __restrict__ b_ih,
        const float* __restrict__ b_hh, float* __restrict__ hn,
        float* __restrict__ cn, float* __restrict__ ds) {
    int t = blockIdx.x * 256 + threadIdx.x;   // 0..16383
    int b = t >> 9, u = t & 511;
    float gi = 0, gf = 0, gg = 0, go = 0;
#pragma unroll
    for (int ks = 0; ks < KSLICES; ++ks) {
        const float* p = partial + (long)ks * (Bsz * Gsz) + (long)b * Gsz;
        gi += p[u]; gf += p[512 + u]; gg += p[1024 + u]; go += p[1536 + u];
    }
    gi += b_ih[u] + b_hh[u];
    gf += b_ih[512 + u] + b_hh[512 + u];
    gg += b_ih[1024 + u] + b_hh[1024 + u];
    go += b_ih[1536 + u] + b_hh[1536 + u];
    float c = sigf(gf) * c0[t] + sigf(gi) * tanhf(gg);
    float h = sigf(go) * tanhf(c);
    cn[t] = c; hn[t] = h; ds[t] = h;
}

// ---------------- K3: bias[b,u] = ds@w_d.T + cov@w_g_w.T + w_g_b ----------------
__global__ __launch_bounds__(256) void k_attnbias(const float* __restrict__ ds,
        const float* __restrict__ w_d, const float* __restrict__ cov,
        const float* __restrict__ w_g_w, const float* __restrict__ w_g_b,
        float* __restrict__ bias) {
    __shared__ _Float16 As[2 * 32 * LDK];
    __shared__ _Float16 Bs[2 * 64 * LDK];
    int tid = threadIdx.x, lane = tid & 31, w = tid >> 5;
    int nt = blockIdx.x;   // 8 blocks x 64 over U
    int tr = (w >> 2) * 16, tc = (w & 3) * 16;
    v8f acc = {};
    gemm_seg(acc, As, Bs, ds,  (long)Usz, w_d   + (long)nt * 64 * Usz, (long)Usz, 0, Usz, tr, tc);
    gemm_seg(acc, As, Bs, cov, (long)Ssz, w_g_w + (long)nt * 64 * Ssz, (long)Ssz, 0, Ssz, tr, tc);
    int n  = nt * 64 + tc + (lane & 15);
    int m0 = tr + ((lane >> 4) << 3);
    float gb = w_g_b[n];
#pragma unroll
    for (int r = 0; r < 8; ++r) bias[(m0 + r) * Usz + n] = acc[r] + gb;
}

// -------- K4: fused feat GEMM + tanh + V_w dot -> raw_scores (feat never hits memory) --------
__global__ __launch_bounds__(256) void k_feat(const float* __restrict__ enc,
        const float* __restrict__ w_e, const float* __restrict__ bias,
        const float* __restrict__ V_w, float* __restrict__ raw_scores) {
    __shared__ _Float16 As[32 * 520];      // 32 rows of (b,s), full K=512
    __shared__ _Float16 Bs[2 * 64 * LDK];
    __shared__ float rowsum2[4 * 32];
    __shared__ float biasS[Usz];
    __shared__ float vwS[Usz];
    int tid = threadIdx.x, lane = tid & 31, w = tid >> 5;
    long rowBase = (long)blockIdx.x * 32;  // over B*S
    int b = (int)(rowBase >> 11);
    {
        StageRegs<32, 512> rA;
        loadQ(rA, enc + rowBase * Usz, (long)Usz, 0);
        storeQ(rA, As, 520);
    }
    for (int i = tid; i < Usz; i += 256) { biasS[i] = bias[b * Usz + i]; vwS[i] = V_w[i]; }
    if (tid < 128) rowsum2[tid] = 0.0f;
    int tr = (w & 1) * 16, tc = (w >> 1) * 16, q = w >> 1;
    for (int nt = 0; nt < 8; ++nt) {
        const float* wnt = w_e + (long)nt * 64 * Usz;
        StageRegs<64, 64> rb;
        loadQ(rb, wnt, (long)Usz, 0);
        storeQ(rb, Bs, LDK);
        v8f acc = {};
        int cur = 0;
        for (int ci = 0; ci < 8; ++ci) {
            __syncthreads();
            int nxt = cur ^ 1;
            bool more = (ci + 1) < 8;
            if (more) loadQ(rb, wnt, (long)Usz, (ci + 1) * 64);
            const _Float16* Bb = Bs + cur * (64 * LDK);
            int kb = ci * 64;
            v16h a0 = ldsFragA(As, 520, tr, kb);
            v16h b0 = ldsFragB(Bb, LDK, tc, 0);
            acc = wmma32(a0, b0, acc);
            v16h a1 = ldsFragA(As, 520, tr, kb + 32);
            v16h b1 = ldsFragB(Bb, LDK, tc, 32);
            acc = wmma32(a1, b1, acc);
            if (more) storeQ(rb, Bs + nxt * (64 * LDK), LDK);
            cur = nxt;
        }
        __syncthreads();
        int u  = nt * 64 + tc + (lane & 15);
        int m0 = tr + ((lane >> 4) << 3);
        float vw = vwS[u], bi = biasS[u];
#pragma unroll
        for (int r = 0; r < 8; ++r) {
            float val = vw * tanhf(acc[r] + bi);
            val += __shfl_xor(val, 1, 32);
            val += __shfl_xor(val, 2, 32);
            val += __shfl_xor(val, 4, 32);
            val += __shfl_xor(val, 8, 32);
            if ((lane & 15) == 0) rowsum2[q * 32 + m0 + r] += val;  // unique (q,row) per wave-half
        }
    }
    __syncthreads();
    if (tid < 32) {
        raw_scores[rowBase + tid] = rowsum2[tid] + rowsum2[32 + tid]
                                  + rowsum2[64 + tid] + rowsum2[96 + tid];
    }
}

// ---------------- K5: softmax over S per batch ----------------
__global__ __launch_bounds__(256) void k_softmax(const float* __restrict__ rs,
        float* __restrict__ attn_out, float* __restrict__ attn_ws) {
    __shared__ float buf[Ssz];
    __shared__ float red[8];
    __shared__ float stat[2];
    int b = blockIdx.x, tid = threadIdx.x, lane = tid & 31, wid = tid >> 5;
    const float* r = rs + (long)b * Ssz;
    float mx = -3.0e38f;
    for (int s = tid; s < Ssz; s += 256) { float v = r[s]; buf[s] = v; mx = fmaxf(mx, v); }
#pragma unroll
    for (int m = 16; m; m >>= 1) mx = fmaxf(mx, __shfl_xor(mx, m, 32));
    if (lane == 0) red[wid] = mx;
    __syncthreads();
    if (tid == 0) { float v = red[0]; for (int i = 1; i < 8; ++i) v = fmaxf(v, red[i]); stat[0] = v; }
    __syncthreads();
    float gmx = stat[0], sum = 0.f;
    for (int s = tid; s < Ssz; s += 256) { float e = __expf(buf[s] - gmx); buf[s] = e; sum += e; }
#pragma unroll
    for (int m = 16; m; m >>= 1) sum += __shfl_xor(sum, m, 32);
    if (lane == 0) red[wid] = sum;
    __syncthreads();
    if (tid == 0) { float v = 0; for (int i = 0; i < 8; ++i) v += red[i]; stat[1] = v; }
    __syncthreads();
    float inv = 1.0f / stat[1];
    for (int s = tid; s < Ssz; s += 256) {
        float a = buf[s] * inv;
        attn_out[(long)b * Ssz + s] = a;
        attn_ws[(long)b * Ssz + s] = a;
    }
}

// ---------------- K6: context = attn @ enc ; s_h, s_s dots ----------------
__global__ __launch_bounds__(512) void k_context(const float* __restrict__ attn,
        const float* __restrict__ enc, const float* __restrict__ ds,
        const float* __restrict__ w_h, const float* __restrict__ w_s,
        float* __restrict__ ctx, float* __restrict__ s_h, float* __restrict__ s_s) {
    __shared__ float aS[Ssz];
    __shared__ float part[4 * Usz];
    __shared__ float red[16];
    int b = blockIdx.x, tid = threadIdx.x, lane = tid & 31, wid = tid >> 5;
    for (int s = tid; s < Ssz; s += 512) aS[s] = attn[(long)b * Ssz + s];
    __syncthreads();
    int c = tid & 127, sl = tid >> 7;              // float4 column, S-slice
    const float* ebase = enc + (long)b * Ssz * Usz + c * 4;
    float4 a4 = {0.f, 0.f, 0.f, 0.f};
    int s0 = sl * 512;
#pragma unroll 4
    for (int s = s0; s < s0 + 512; ++s) {
        float a = aS[s];
        float4 v = *(const float4*)(ebase + (long)s * Usz);
        a4.x = fmaf(a, v.x, a4.x);
        a4.y = fmaf(a, v.y, a4.y);
        a4.z = fmaf(a, v.z, a4.z);
        a4.w = fmaf(a, v.w, a4.w);
    }
    float* pp = part + sl * Usz + c * 4;
    pp[0] = a4.x; pp[1] = a4.y; pp[2] = a4.z; pp[3] = a4.w;
    __syncthreads();
    float cv = part[tid] + part[Usz + tid] + part[2 * Usz + tid] + part[3 * Usz + tid];
    ctx[b * Usz + tid] = cv;
    float vh = cv * w_h[tid];
#pragma unroll
    for (int m = 16; m; m >>= 1) vh += __shfl_xor(vh, m, 32);
    if (lane == 0) red[wid] = vh;
    __syncthreads();
    if (tid == 0) { float v = 0; for (int i = 0; i < 16; ++i) v += red[i]; s_h[b] = v; }
    __syncthreads();
    float vs = ds[b * Usz + tid] * w_s[tid];
#pragma unroll
    for (int m = 16; m; m >>= 1) vs += __shfl_xor(vs, m, 32);
    if (lane == 0) red[wid] = vs;
    __syncthreads();
    if (tid == 0) { float v = 0; for (int i = 0; i < 16; ++i) v += red[i]; s_s[b] = v; }
}

// ---------------- s_x[b] = new_tokens[b]·w_x_w ----------------
__global__ __launch_bounds__(256) void k_sx(const float* __restrict__ x,
        const float* __restrict__ wxw, float* __restrict__ s_x) {
    __shared__ float red[8];
    int b = blockIdx.x, tid = threadIdx.x, lane = tid & 31, wid = tid >> 5;
    const float4* x4 = (const float4*)(x + (long)b * Vsz);
    const float4* w4 = (const float4*)wxw;
    float v = 0.f;
    for (int k = tid; k < Vsz / 4; k += 256) {
        float4 a = x4[k], ww = w4[k];
        v = fmaf(a.x, ww.x, v); v = fmaf(a.y, ww.y, v);
        v = fmaf(a.z, ww.z, v); v = fmaf(a.w, ww.w, v);
    }
#pragma unroll
    for (int m = 16; m; m >>= 1) v += __shfl_xor(v, m, 32);
    if (lane == 0) red[wid] = v;
    __syncthreads();
    if (tid == 0) { float t = 0; for (int i = 0; i < 8; ++i) t += red[i]; s_x[b] = t; }
}

__global__ void k_pgen(const float* __restrict__ s_h, const float* __restrict__ s_s,
                       const float* __restrict__ s_x, const float* __restrict__ w_x_b,
                       float* __restrict__ pg) {
    int b = threadIdx.x;
    if (b < Bsz) pg[b] = sigf(s_h[b] + s_s[b] + s_x[b] + w_x_b[0]);
}

// ---------------- K_r: r = [ds|ctx] @ w_c.T ----------------
__global__ __launch_bounds__(256) void k_rvec(const float* __restrict__ ds,
        const float* __restrict__ ctx, const float* __restrict__ w_c,
        float* __restrict__ rvec) {
    __shared__ _Float16 As[2 * 32 * LDK];
    __shared__ _Float16 Bs[2 * 64 * LDK];
    int tid = threadIdx.x, lane = tid & 31, w = tid >> 5;
    int nt = blockIdx.x;   // 8 blocks x 64 over U
    int tr = (w >> 2) * 16, tc = (w & 3) * 16;
    v8f acc = {};
    gemm_seg(acc, As, Bs, ds,  (long)Usz, w_c + (long)nt * 64 * 1024,       1024L, 0, Usz, tr, tc);
    gemm_seg(acc, As, Bs, ctx, (long)Usz, w_c + (long)nt * 64 * 1024 + 512, 1024L, 0, Usz, tr, tc);
    int n  = nt * 64 + tc + (lane & 15);
    int m0 = tr + ((lane >> 4) << 3);
#pragma unroll
    for (int r = 0; r < 8; ++r) rvec[(m0 + r) * Usz + n] = acc[r];
}

// ---------------- K7: logits = p_gen * (r @ f_c_w.T + f_c_b) ----------------
__global__ __launch_bounds__(256) void k_logits(const float* __restrict__ rvec,
        const float* __restrict__ f_c_w, const float* __restrict__ f_c_b,
        const float* __restrict__ pg, float* __restrict__ out) {
    __shared__ _Float16 As[32 * 520];
    __shared__ _Float16 Bs[2 * 64 * LDK];
    int tid = threadIdx.x, lane = tid & 31, w = tid >> 5;
    int vBase = blockIdx.x * 64;   // 500 blocks over V
    {
        StageRegs<32, 512> rA;
        loadQ(rA, rvec, (long)Usz, 0);
        storeQ(rA, As, 520);
    }
    const float* fw = f_c_w + (long)vBase * Usz;
    StageRegs<64, 64> rb;
    loadQ(rb, fw, (long)Usz, 0);
    storeQ(rb, Bs, LDK);
    int tr = (w >> 2) * 16, tc = (w & 3) * 16;
    v8f acc = {};
    int cur = 0;
    for (int ci = 0; ci < 8; ++ci) {
        __syncthreads();
        int nxt = cur ^ 1;
        bool more = (ci + 1) < 8;
        if (more) loadQ(rb, fw, (long)Usz, (ci + 1) * 64);
        const _Float16* Bb = Bs + cur * (64 * LDK);
        int kb = ci * 64;
        v16h a0 = ldsFragA(As, 520, tr, kb);
        v16h b0 = ldsFragB(Bb, LDK, tc, 0);
        acc = wmma32(a0, b0, acc);
        v16h a1 = ldsFragA(As, 520, tr, kb + 32);
        v16h b1 = ldsFragB(Bb, LDK, tc, 32);
        acc = wmma32(a1, b1, acc);
        if (more) storeQ(rb, Bs + nxt * (64 * LDK), LDK);
        cur = nxt;
    }
    int vcol = vBase + tc + (lane & 15);
    int m0   = tr + ((lane >> 4) << 3);
    float fb = f_c_b[vcol];
#pragma unroll
    for (int r = 0; r < 8; ++r) {
        int bb = m0 + r;
        out[(long)bb * Vsz + vcol] = pg[bb] * (acc[r] + fb);
    }
}

__global__ __launch_bounds__(256) void k_copy(const float* __restrict__ rs,
        const float* __restrict__ pg, float* __restrict__ out) {
    int t = blockIdx.x * 256 + threadIdx.x;   // 65536
    int b = t >> 11;
    out[t] = rs[t] * (1.0f - pg[b]);
}

extern "C" void kernel_launch(void* const* d_in, const int* in_sizes, int n_in,
                              void* d_out, int out_size, void* d_ws, size_t ws_size,
                              hipStream_t stream) {
    (void)in_sizes; (void)n_in; (void)out_size; (void)ws_size;
    const float* new_tokens = (const float*)d_in[0];
    const float* enc   = (const float*)d_in[1];
    const float* h0    = (const float*)d_in[2];
    const float* c0    = (const float*)d_in[3];
    const float* cov   = (const float*)d_in[4];
    const float* W_ih  = (const float*)d_in[5];
    const float* W_hh  = (const float*)d_in[6];
    const float* b_ih  = (const float*)d_in[7];
    const float* b_hh  = (const float*)d_in[8];
    const float* w_d   = (const float*)d_in[9];
    const float* w_e   = (const float*)d_in[10];
    const float* w_g_w = (const float*)d_in[11];
    const float* w_g_b = (const float*)d_in[12];
    const float* V_w   = (const float*)d_in[13];
    const float* w_c   = (const float*)d_in[14];
    const float* f_c_w = (const float*)d_in[15];
    const float* f_c_b = (const float*)d_in[16];
    const float* w_h   = (const float*)d_in[17];
    const float* w_s   = (const float*)d_in[18];
    const float* w_x_w = (const float*)d_in[19];
    const float* w_x_b = (const float*)d_in[20];

    float* out      = (float*)d_out;
    float* o_logits = out;                         // 32*32000
    float* o_attn   = out + 1024000;               // 32*2048
    float* o_h      = out + 1024000 + 65536;       // 32*512
    float* o_c      = o_h + 16384;                 // 32*512
    float* o_copy   = o_c + 16384;                 // 32*2048

    float* ws     = (float*)d_ws;
    float* w_part = ws;                       // KSLICES*32*2048 = 655360
    float* w_ds   = w_part + KSLICES * Bsz * Gsz;
    float* w_bias = w_ds + 16384;
    float* w_rs   = w_bias + 16384;           // 65536
    float* w_at   = w_rs + 65536;             // 65536
    float* w_ctx  = w_at + 65536;
    float* w_r    = w_ctx + 16384;
    float* w_sx   = w_r + 16384;
    float* w_sh   = w_sx + 32;
    float* w_ss   = w_sh + 32;
    float* w_pg   = w_ss + 32;

    k_gates   <<<dim3(32, KSLICES), 256, 0, stream>>>(new_tokens, W_ih, h0, W_hh, w_part);
    k_sx      <<<32,  256, 0, stream>>>(new_tokens, w_x_w, w_sx);
    k_lstm    <<<64,  256, 0, stream>>>(w_part, c0, b_ih, b_hh, o_h, o_c, w_ds);
    k_attnbias<<<8,   256, 0, stream>>>(w_ds, w_d, cov, w_g_w, w_g_b, w_bias);
    k_feat    <<<2048,256, 0, stream>>>(enc, w_e, w_bias, V_w, w_rs);
    k_softmax <<<32,  256, 0, stream>>>(w_rs, o_attn, w_at);
    k_context <<<32,  512, 0, stream>>>(w_at, enc, w_ds, w_h, w_s, w_ctx, w_sh, w_ss);
    k_pgen    <<<1,    32, 0, stream>>>(w_sh, w_ss, w_sx, w_x_b, w_pg);
    k_rvec    <<<8,   256, 0, stream>>>(w_ds, w_ctx, w_c, w_r);
    k_logits  <<<500, 256, 0, stream>>>(w_r, f_c_w, f_c_b, w_pg, o_logits);
    k_copy    <<<256, 256, 0, stream>>>(w_rs, w_pg, o_copy);
}
